// SeqModelBGRU_HC_30511447671465
// MI455X (gfx1250) — matrix operand order repaired and checked by reading, not compiled, not run on hardware
//
#include <hip/hip_runtime.h>
#include <hip/hip_bf16.h>

// ---------------------------------------------------------------------------
// CDNA5 (gfx1250) mixed-precision implementation of the BiGRU + top-k model.
// Heavy GEMMs use v_wmma_f32_16x16x32_f16; GEMM LDS staging uses the CDNA5
// async global->LDS data path (ASYNCcnt) when the builtin is available.
// ---------------------------------------------------------------------------

typedef _Float16 h8   __attribute__((ext_vector_type(8)));
typedef _Float16 v16h __attribute__((ext_vector_type(16)));
typedef float    v8f  __attribute__((ext_vector_type(8)));
typedef int      v4i  __attribute__((ext_vector_type(4)));

#define AS1 __attribute__((address_space(1)))
#define AS3 __attribute__((address_space(3)))

#if __has_builtin(__builtin_amdgcn_global_load_async_to_lds_b128)
#define HAVE_ASYNC_LDS 1
#endif

static constexpr int Bn = 128, Tn = 256, Dn = 512, Hn = 256;
static constexpr int G3 = 3 * Hn;      // 768
static constexpr int Mrows = Bn * Tn;  // 32768

__device__ inline v8f vzero8() {
  v8f z;
#pragma unroll
  for (int i = 0; i < 8; ++i) z[i] = 0.0f;
  return z;
}

// 16-byte global -> LDS copy: async (ASYNCcnt-tracked, no VGPR staging) when
// available, else a plain load+store (same barrier schedule is correct).
__device__ inline void cp16_g2l(const _Float16* g, _Float16* l) {
#if defined(HAVE_ASYNC_LDS)
  __builtin_amdgcn_global_load_async_to_lds_b128((AS1 v4i*)(g), (AS3 v4i*)(l),
                                                 0, 0);
#else
  *(h8*)l = *(const h8*)g;
#endif
}

__device__ inline void wait_async0() {
#if defined(HAVE_ASYNC_LDS)
#if __has_builtin(__builtin_amdgcn_s_wait_asynccnt)
  __builtin_amdgcn_s_wait_asynccnt(0);
#else
  asm volatile("s_wait_asynccnt 0x0" ::: "memory");
#endif
#endif
}

// A-matrix fragment (16x32 f16, MxK) from a row-major f16 tile, row stride ld.
// Layout (ISA 7.12.2): lane -> M = lane&15; half e<8 -> K = e + 8*(lane>=16);
// e>=8 -> K = 16 + (e-8) + 8*(lane>=16).
__device__ inline v16h frag_a(const _Float16* tile, int ld) {
  int l = threadIdx.x & 31;
  const _Float16* base = tile + (l & 15) * ld + ((l >> 4) & 1) * 8;
  h8 lo = *(const h8*)(base);
  h8 hi = *(const h8*)(base + 16);
  v16h r;
#pragma unroll
  for (int i = 0; i < 8; ++i) { r[i] = lo[i]; r[i + 8] = hi[i]; }
  return r;
}

// B-matrix fragment (32x16 f16, KxN) where B[k][n] = W[n][k], W row-major with
// row stride ld. Layout: lane -> N = lane&15; half e<8 -> K = e + 8*(lane>=16);
// e>=8 -> K = 16 + (e-8) + 8*(lane>=16).
__device__ inline v16h frag_b(const _Float16* wrows, int ld) {
  int l = threadIdx.x & 31;
  const _Float16* base = wrows + (l & 15) * ld + ((l >> 4) & 1) * 8;
  h8 lo = *(const h8*)(base);
  h8 hi = *(const h8*)(base + 16);
  v16h r;
#pragma unroll
  for (int i = 0; i < 8; ++i) { r[i] = lo[i]; r[i + 8] = hi[i]; }
  return r;
}

__device__ inline v8f wmma16(v16h a, v16h b, v8f c) {
  return __builtin_amdgcn_wmma_f32_16x16x32_f16(false, a, false, b, (short)0, c,
                                                false, false);
}

// ---------------------------------------------------------------------------
// Kernel 1: f32 -> f16 conversion
// ---------------------------------------------------------------------------
__global__ void cvt_f32_f16(const float* __restrict__ src,
                            _Float16* __restrict__ dst, int n) {
  int i = blockIdx.x * blockDim.x + threadIdx.x;
  if (i < n) dst[i] = (_Float16)src[i];
}

// ---------------------------------------------------------------------------
// Kernel 2: xg[dir] = feats(f16) @ w_ih[dir]^T + b_ih[dir]   (M=32768,K=512,N=768)
// 128x128 tile / workgroup, 8 waves as 4(M)x2(N), LDS double-buffered with
// async global->LDS staging (one barrier per K stage).
// ---------------------------------------------------------------------------
__global__ __launch_bounds__(256) void xg_gemm(
    const _Float16* __restrict__ A, const _Float16* __restrict__ Wf,
    const _Float16* __restrict__ Wb, const float* __restrict__ bias_f,
    const float* __restrict__ bias_b, float* __restrict__ xg) {
  constexpr int LD = 40;  // padded half-stride (80B): 16B aligned, bank-clean
  __shared__ _Float16 As[2][128 * LD];
  __shared__ _Float16 Bs[2][128 * LD];

  const int mt = blockIdx.x, nt = blockIdx.y, dir = blockIdx.z;
  const _Float16* W = dir ? Wb : Wf;
  const float* bias = dir ? bias_b : bias_f;
  float* out = xg + (size_t)dir * Mrows * G3;

  const int tid = threadIdx.x, w = tid >> 5, lane = tid & 31;
  const int waveM = w & 3, waveN = w >> 2;
  const int m0 = mt * 128, n0 = nt * 128;
  const int lr = tid >> 1, lo = (tid & 1) * 16;
  const _Float16* gA = A + (size_t)(m0 + lr) * Dn + lo;
  const _Float16* gB = W + (size_t)(n0 + lr) * Dn + lo;
  _Float16* lA0 = &As[0][lr * LD + lo];
  _Float16* lB0 = &Bs[0][lr * LD + lo];
  _Float16* lA1 = &As[1][lr * LD + lo];
  _Float16* lB1 = &Bs[1][lr * LD + lo];

  // stage 0
  cp16_g2l(gA, lA0);      cp16_g2l(gA + 8, lA0 + 8);
  cp16_g2l(gB, lB0);      cp16_g2l(gB + 8, lB0 + 8);
  wait_async0();
  __syncthreads();

  v8f acc[2][4];
#pragma unroll
  for (int i = 0; i < 2; ++i)
#pragma unroll
    for (int j = 0; j < 4; ++j) acc[i][j] = vzero8();

  for (int kt = 0; kt < 16; ++kt) {
    const int cur = kt & 1;
    if (kt < 15) {
      // async-stage K slice kt+1 into the other buffer (read last at kt-1,
      // protected by the barrier that ended stage kt-1)
      _Float16* la = cur ? lA0 : lA1;
      _Float16* lb = cur ? lB0 : lB1;
      const _Float16* pa = gA + (kt + 1) * 32;
      const _Float16* pb = gB + (kt + 1) * 32;
      cp16_g2l(pa, la);      cp16_g2l(pa + 8, la + 8);
      cp16_g2l(pb, lb);      cp16_g2l(pb + 8, lb + 8);
      if (kt < 14) {  // global_prefetch_b8 of next-next K slice
        __builtin_prefetch(gA + (kt + 2) * 32, 0, 1);
        __builtin_prefetch(gB + (kt + 2) * 32, 0, 1);
      }
    }
    v16h a0 = frag_a(&As[cur][(waveM * 32) * LD], LD);
    v16h a1 = frag_a(&As[cur][(waveM * 32 + 16) * LD], LD);
#pragma unroll
    for (int n = 0; n < 4; ++n) {
      v16h bf = frag_b(&Bs[cur][(waveN * 64 + n * 16) * LD], LD);
      acc[0][n] = wmma16(a0, bf, acc[0][n]);
      acc[1][n] = wmma16(a1, bf, acc[1][n]);
    }
    if (kt < 15) {
      wait_async0();
      __syncthreads();
    }
  }

  const int cl = lane & 15, hi2 = lane >> 4;
#pragma unroll
  for (int sub = 0; sub < 2; ++sub) {
    const int rowb = m0 + waveM * 32 + sub * 16 + hi2 * 8;
#pragma unroll
    for (int n = 0; n < 4; ++n) {
      const int col = n0 + waveN * 64 + n * 16 + cl;
      const float bb = bias[col];
#pragma unroll
      for (int r = 0; r < 8; ++r)
        out[(size_t)(rowb + r) * G3 + col] = acc[sub][n][r] + bb;
    }
  }
}

// ---------------------------------------------------------------------------
// Kernel 3: GRU time scan. One workgroup = (direction, 16 batch rows).
// Per step: hg = h(f16,LDS) @ w_hh^T via WMMA (8 waves x 6 N-tiles x 8 K-tiles)
// then gate math in f32 with length masking. Backward dir scans reversed time.
// ---------------------------------------------------------------------------
__global__ __launch_bounds__(256) void gru_scan(
    const float* __restrict__ xg_all, const _Float16* __restrict__ whh_f,
    const _Float16* __restrict__ whh_b, const float* __restrict__ bhh_f,
    const float* __restrict__ bhh_b, const int* __restrict__ lengths,
    float* __restrict__ out_f32, _Float16* __restrict__ out_f16) {
  const int dir = blockIdx.x >> 3;
  const int b0 = (blockIdx.x & 7) * 16;
  const _Float16* Whh = dir ? whh_b : whh_f;
  const float* bhh = dir ? bhh_b : bhh_f;
  const float* xg = xg_all + (size_t)dir * Mrows * G3;

  __shared__ _Float16 hA[16 * Hn];   // f16 state for WMMA A fragments
  __shared__ float    hS[16 * Hn];   // f32 master state
  __shared__ float    hg[16 * G3];   // per-step gate pre-activations

  const int tid = threadIdx.x, w = tid >> 5, lane = tid & 31;
  for (int i = tid; i < 16 * Hn; i += 256) { hA[i] = (_Float16)0.0f; hS[i] = 0.0f; }
  __syncthreads();

  const int erow = tid >> 4;        // elementwise: batch row 0..15
  const int ecol = tid & 15;        // elementwise: col lane 0..15
  const int b = b0 + erow;
  const int len = lengths[b];

  for (int s = 0; s < Tn; ++s) {
    // ---- matmul phase: hg = h @ Whh^T (all threads, uniform flow) ----
    v16h af[8];
#pragma unroll
    for (int kt = 0; kt < 8; ++kt) af[kt] = frag_a(&hA[kt * 32], Hn);
#pragma unroll
    for (int nt = 0; nt < 6; ++nt) {
      const int ncol = (w * 6 + nt) * 16;
      v8f acc = vzero8();
#pragma unroll
      for (int kt = 0; kt < 8; ++kt) {
        v16h bf = frag_b(Whh + (size_t)ncol * Hn + kt * 32, Hn);
        acc = wmma16(af[kt], bf, acc);
      }
      const int c = ncol + (lane & 15);
      const int rb = (lane >> 4) * 8;
#pragma unroll
      for (int r = 0; r < 8; ++r) hg[(rb + r) * G3 + c] = acc[r];
    }
    __syncthreads();

    // ---- elementwise gate phase ----
    const bool valid = s < len;
    const int t = dir ? (len - 1 - s) : s;
#pragma unroll 4
    for (int j = 0; j < 16; ++j) {
      const int col = ecol + j * 16;
      if (valid) {
        const size_t xb = ((size_t)b * Tn + t) * G3;
        const float xr = xg[xb + col];
        const float xz = xg[xb + Hn + col];
        const float xn = xg[xb + 2 * Hn + col];
        const float hr = hg[erow * G3 + col] + bhh[col];
        const float hz = hg[erow * G3 + Hn + col] + bhh[Hn + col];
        const float hn = hg[erow * G3 + 2 * Hn + col] + bhh[2 * Hn + col];
        const float hold = hS[erow * Hn + col];
        const float rg = 1.0f / (1.0f + expf(-(xr + hr)));
        const float zg = 1.0f / (1.0f + expf(-(xz + hz)));
        const float ng = tanhf(xn + rg * hn);
        const float hnew = (1.0f - zg) * ng + zg * hold;
        hS[erow * Hn + col] = hnew;
        hA[erow * Hn + col] = (_Float16)hnew;
        const size_t o = ((size_t)b * Tn + t) * (2 * Hn) + dir * Hn + col;
        out_f32[o] = hnew;
        out_f16[o] = (_Float16)hnew;
      } else {  // padded position: step index s itself is a pad slot
        const size_t o = ((size_t)b * Tn + s) * (2 * Hn) + dir * Hn + col;
        out_f32[o] = 0.0f;
        out_f16[o] = (_Float16)0.0f;
      }
    }
    __syncthreads();
  }
}

// ---------------------------------------------------------------------------
// Kernel 4: score MLP: scores = relu(out @ w1^T + b1) @ w2^T + b2
// 128 rows / workgroup, 8 waves x 16 rows, WMMA K=512 N=64, fused reduction.
// ---------------------------------------------------------------------------
__global__ __launch_bounds__(256) void mlp_scores(
    const _Float16* __restrict__ X, const _Float16* __restrict__ W1h,
    const float* __restrict__ b1, const float* __restrict__ w2,
    const float* __restrict__ b2, float* __restrict__ scores) {
  const int tid = threadIdx.x, w = tid >> 5, lane = tid & 31;
  const int m0 = blockIdx.x * 128 + w * 16;

  v8f acc[4];
#pragma unroll
  for (int n = 0; n < 4; ++n) acc[n] = vzero8();

  for (int kt = 0; kt < 16; ++kt) {
    v16h af = frag_a(X + (size_t)m0 * (2 * Hn) + kt * 32, 2 * Hn);
#pragma unroll
    for (int nt = 0; nt < 4; ++nt) {
      v16h bf = frag_b(W1h + (size_t)(nt * 16) * (2 * Hn) + kt * 32, 2 * Hn);
      acc[nt] = wmma16(af, bf, acc[nt]);
    }
  }

  __shared__ float red[128][16];
  const int nloc = lane & 15, hi2 = lane >> 4;
  float part[8];
#pragma unroll
  for (int r = 0; r < 8; ++r) part[r] = 0.0f;
#pragma unroll
  for (int nt = 0; nt < 4; ++nt) {
    const int n = nt * 16 + nloc;
    const float wb = w2[n], bb = b1[n];
#pragma unroll
    for (int r = 0; r < 8; ++r) {
      float v = acc[nt][r] + bb;
      v = v > 0.0f ? v : 0.0f;
      part[r] += v * wb;
    }
  }
#pragma unroll
  for (int r = 0; r < 8; ++r) red[w * 16 + hi2 * 8 + r][nloc] = part[r];
  __syncthreads();
  if (tid < 128) {
    float sum = 0.0f;
#pragma unroll
    for (int i = 0; i < 16; ++i) sum += red[tid][i];
    scores[(size_t)blockIdx.x * 128 + tid] = sum + b2[0];
  }
}

// ---------------------------------------------------------------------------
// Kernel 5: masked softmax + top-3 + normalize + fallback + weighted sum + heads
// One workgroup per batch row.
// ---------------------------------------------------------------------------
__global__ __launch_bounds__(256) void attn_heads(
    const float* __restrict__ scores, const float* __restrict__ out,
    const int* __restrict__ lengths, const float* __restrict__ temp_p,
    const float* __restrict__ w_tens, const float* __restrict__ b_tens,
    const float* __restrict__ w_ones, const float* __restrict__ b_ones,
    float* __restrict__ d_out) {
  const int b = blockIdx.x, t = threadIdx.x;
  __shared__ float sm[256], probs[256], lv[256], attn[256], seq[2 * Hn];
  __shared__ int li[256], topi_s[3];
  __shared__ float topv_s[3];

  const int len = lengths[b];
  float temp = temp_p[0];
  temp = fminf(fmaxf(temp, 0.001f), 10.0f);
  const bool valid = t < len;
  const float logit = valid ? scores[(size_t)b * Tn + t] / temp : -1e30f;

  sm[t] = logit;
  __syncthreads();
  for (int s2 = 128; s2 > 0; s2 >>= 1) {
    if (t < s2) sm[t] = fmaxf(sm[t], sm[t + s2]);
    __syncthreads();
  }
  const float mx = sm[0];
  __syncthreads();
  const float e = valid ? expf(logit - mx) : 0.0f;
  sm[t] = e;
  __syncthreads();
  for (int s2 = 128; s2 > 0; s2 >>= 1) {
    if (t < s2) sm[t] += sm[t + s2];
    __syncthreads();
  }
  const float ssum = sm[0];
  __syncthreads();
  const float p = e / ssum;
  probs[t] = p;
  attn[t] = 0.0f;
  __syncthreads();

  // top-3 via 3 argmax reductions (tie -> lowest index, matches lax.top_k)
  for (int k = 0; k < 3; ++k) {
    lv[t] = probs[t];
    li[t] = t;
    __syncthreads();
    for (int s2 = 128; s2 > 0; s2 >>= 1) {
      if (t < s2) {
        const float v2 = lv[t + s2];
        const int i2 = li[t + s2];
        if (v2 > lv[t] || (v2 == lv[t] && i2 < li[t])) { lv[t] = v2; li[t] = i2; }
      }
      __syncthreads();
    }
    if (t == 0) { topv_s[k] = lv[0]; topi_s[k] = li[0]; }
    __syncthreads();
    if (t == topi_s[k]) probs[t] = -1.0f;
    __syncthreads();
  }

  const int kact = len < 3 ? len : 3;
  float vals[3], vsum = 0.0f;
#pragma unroll
  for (int k = 0; k < 3; ++k) {
    vals[k] = (k < kact) ? topv_s[k] : 0.0f;
    vsum += vals[k];
  }
  if (vsum > 1e-8f) {
#pragma unroll
    for (int k = 0; k < 3; ++k)
      if (k < kact && t == topi_s[k]) attn[t] = vals[k] / fmaxf(vsum, 1e-8f);
  } else {
    attn[t] = valid ? 1.0f / ((float)len + 1e-8f) : 0.0f;
  }
  __syncthreads();

  // weighted sum over T (attn is sparse; scalar branch skips zero weights)
  float a0 = 0.0f, a1 = 0.0f;
  for (int tt = 0; tt < Tn; ++tt) {
    const float a = attn[tt];
    if (a != 0.0f) {
      const float* orow = out + ((size_t)b * Tn + tt) * (2 * Hn);
      a0 += a * orow[t];
      a1 += a * orow[Hn + t];
    }
  }
  seq[t] = a0;
  seq[Hn + t] = a1;
  __syncthreads();

  if (t < 21) {
    const float* wrow;
    float bias;
    float* dst;
    if (t < 11) {
      wrow = w_tens + (size_t)t * (2 * Hn);
      bias = b_tens[t];
      dst = d_out + (size_t)b * 11 + t;
    } else {
      const int o = t - 11;
      wrow = w_ones + (size_t)o * (2 * Hn);
      bias = b_ones[o];
      dst = d_out + (size_t)Bn * 11 + (size_t)b * 10 + o;
    }
    float sacc = bias;
    for (int i = 0; i < 2 * Hn; ++i) sacc += wrow[i] * seq[i];
    *dst = sacc;
  }
}

// ---------------------------------------------------------------------------
// Host launch
// ---------------------------------------------------------------------------
extern "C" void kernel_launch(void* const* d_in, const int* in_sizes, int n_in,
                              void* d_out, int out_size, void* d_ws,
                              size_t ws_size, hipStream_t stream) {
  (void)in_sizes; (void)n_in; (void)out_size; (void)ws_size;

  const float* feats   = (const float*)d_in[0];
  const int*   lengths = (const int*)d_in[1];
  const float* temp    = (const float*)d_in[2];
  const float* w_ih_f  = (const float*)d_in[3];
  const float* w_hh_f  = (const float*)d_in[4];
  const float* b_ih_f  = (const float*)d_in[5];
  const float* b_hh_f  = (const float*)d_in[6];
  const float* w_ih_b  = (const float*)d_in[7];
  const float* w_hh_b  = (const float*)d_in[8];
  const float* b_ih_b  = (const float*)d_in[9];
  const float* b_hh_b  = (const float*)d_in[10];
  const float* w1      = (const float*)d_in[11];
  const float* b1      = (const float*)d_in[12];
  const float* w2      = (const float*)d_in[13];
  const float* b2      = (const float*)d_in[14];
  const float* w_tens  = (const float*)d_in[15];
  const float* b_tens  = (const float*)d_in[16];
  const float* w_ones  = (const float*)d_in[17];
  const float* b_ones  = (const float*)d_in[18];

  constexpr size_t SZ_FEATS16 = (size_t)Bn * Tn * Dn * 2;       // 32 MB
  constexpr size_t SZ_WIH16   = (size_t)G3 * Dn * 2;            // 768 KB
  constexpr size_t SZ_WHH16   = (size_t)G3 * Hn * 2;            // 384 KB
  constexpr size_t SZ_W116    = (size_t)64 * 2 * Hn * 2;        // 64 KB
  constexpr size_t SZ_XG      = (size_t)2 * Mrows * G3 * 4;     // 192 MB
  constexpr size_t SZ_OUT32   = (size_t)Mrows * 2 * Hn * 4;     // 64 MB
  constexpr size_t SZ_OUT16   = (size_t)Mrows * 2 * Hn * 2;     // 32 MB

  char* ws = (char*)d_ws;
  size_t off = 0;
  _Float16* feats16 = (_Float16*)(ws + off); off += SZ_FEATS16;
  _Float16* wihf16  = (_Float16*)(ws + off); off += SZ_WIH16;
  _Float16* wihb16  = (_Float16*)(ws + off); off += SZ_WIH16;
  _Float16* whhf16  = (_Float16*)(ws + off); off += SZ_WHH16;
  _Float16* whhb16  = (_Float16*)(ws + off); off += SZ_WHH16;
  _Float16* w116    = (_Float16*)(ws + off); off += SZ_W116;
  float*    xg      = (float*)(ws + off);    off += SZ_XG;
  float*    out32   = (float*)(ws + off);    off += SZ_OUT32;
  _Float16* out16   = (_Float16*)(ws + off); off += SZ_OUT16;
  float*    scores  = (float*)(ws + off);    off += (size_t)Mrows * 4;

  auto cvt = [&](const float* s, _Float16* d, int n) {
    cvt_f32_f16<<<(n + 255) / 256, 256, 0, stream>>>(s, d, n);
  };
  cvt(feats, feats16, Bn * Tn * Dn);
  cvt(w_ih_f, wihf16, G3 * Dn);
  cvt(w_ih_b, wihb16, G3 * Dn);
  cvt(w_hh_f, whhf16, G3 * Hn);
  cvt(w_hh_b, whhb16, G3 * Hn);
  cvt(w1, w116, 64 * 2 * Hn);

  xg_gemm<<<dim3(Mrows / 128, G3 / 128, 2), 256, 0, stream>>>(
      feats16, wihf16, wihb16, b_ih_f, b_ih_b, xg);

  gru_scan<<<16, 256, 0, stream>>>(xg, whhf16, whhb16, b_hh_f, b_hh_b, lengths,
                                   out32, out16);

  mlp_scores<<<Mrows / 128, 256, 0, stream>>>(out16, w116, b1, w2, b2, scores);

  attn_heads<<<Bn, 256, 0, stream>>>(scores, out32, lengths, temp, w_tens,
                                     b_tens, w_ones, b_ones, (float*)d_out);
}
